// HungarianMatcherDynamicK_84859963835117
// MI455X (gfx1250) — compile-verified, matrix operand order checked by reading, not run on hardware
//
#include <hip/hip_runtime.h>

// ---------------- problem constants ----------------
#define BB   32
#define QQ   4000
#define GG   100
#define C1N  4
#define C2N  32
#define C3N  8
#define NCLS 44          // C1+C2+C3
#define KPAD 64          // padded class dim for 2x K=32 WMMA steps
#define QTILE 128        // queries per block in cost kernel (8 waves * 16 rows)
#define GTILES 7         // ceil(100/16)

typedef __attribute__((ext_vector_type(16))) _Float16 v16h;
typedef __attribute__((ext_vector_type(8)))  float    v8f;

union AFrag { v16h v; _Float16 h[16]; };
union CFrag { v8f  v; float    f[8];  };

// ============================================================
// Kernel 1: fused cost matrix.  grid = (ceil(Q/128), B), block = 256 (8 waves)
// Class cost via v_wmma_f32_16x16x32_f16 (F[q,c] x one-hot(labels)), rest VALU.
// ============================================================
__global__ __launch_bounds__(256) void cost_kernel(
    const float* __restrict__ pl1, const float* __restrict__ pl2,
    const float* __restrict__ pl3, const float* __restrict__ pbox,
    const float* __restrict__ gbox, const float* __restrict__ imgsz,
    const int* __restrict__ lab1, const int* __restrict__ lab2,
    const int* __restrict__ lab3, float* __restrict__ costOut)
{
  __shared__ _Float16 F_lds[QTILE * KPAD];   // 16 KB focal table (f16)
  __shared__ float    pb_lds[QTILE * 4];     // pred boxes for this tile
  __shared__ float    gt_lds[GG * 4];        // all gt boxes
  __shared__ short    gc1[GG], gc2[GG], gc3[GG];
  __shared__ float    fg_pen[QTILE];         // 0 or 10000 per query
  __shared__ float    isz[4];

  const int b     = blockIdx.y;
  const int qbase = blockIdx.x * QTILE;
  const int tid   = threadIdx.x;

  // ---- stage phase ----
  for (int i = tid; i < QTILE * 4; i += 256) {
    int q = qbase + (i >> 2);
    pb_lds[i] = (q < QQ) ? pbox[(size_t)b * QQ * 4 + (size_t)q * 4 + (i & 3)] : 0.0f;
  }
  for (int i = tid; i < GG * 4; i += 256)
    gt_lds[i] = gbox[(size_t)b * GG * 4 + i];
  for (int g = tid; g < GG; g += 256) {
    gc1[g] = (short)lab1[b * GG + g];
    gc2[g] = (short)(C1N + lab2[b * GG + g]);
    gc3[g] = (short)(C1N + C2N + lab3[b * GG + g]);
  }
  if (tid < 4) isz[tid] = imgsz[b * 4 + tid];
  for (int i = tid; i < QTILE * KPAD; i += 256) {
    int r = i >> 6, c = i & 63;
    int q = qbase + r;
    float F = 0.0f;
    if (c < NCLS && q < QQ) {
      float x;
      if (c < C1N)           x = pl1[(size_t)b * QQ * C1N + (size_t)q * C1N + c];
      else if (c < C1N + C2N) x = pl2[(size_t)b * QQ * C2N + (size_t)q * C2N + (c - C1N)];
      else                   x = pl3[(size_t)b * QQ * C3N + (size_t)q * C3N + (c - C1N - C2N)];
      float p   = 1.0f / (1.0f + __expf(-x));
      float pos = 0.25f * (1.0f - p) * (1.0f - p) * (-__logf(p + 1e-8f));
      float neg = 0.75f * p * p * (-__logf(1.0f - p + 1e-8f));
      F = pos - neg;
    }
    F_lds[i] = (_Float16)F;
  }
  __syncthreads();

  // ---- per-query foreground penalty (any over ALL gts) ----
  for (int r = tid; r < QTILE; r += 256) {
    float x0 = pb_lds[r*4+0], y0 = pb_lds[r*4+1], x1 = pb_lds[r*4+2], y1 = pb_lds[r*4+3];
    float ax = 0.5f * (x0 + x1), ay = 0.5f * (y0 + y1);
    bool anyBox = false, anyCtr = false;
    for (int g = 0; g < GG; ++g) {
      float X0 = gt_lds[g*4+0], Y0 = gt_lds[g*4+1], X1 = gt_lds[g*4+2], Y1 = gt_lds[g*4+3];
      anyBox |= (ax > X0) && (ax < X1) && (ay > Y0) && (ay < Y1);
      float w = X1 - X0, h = Y1 - Y0, cx = 0.5f * (X0 + X1), cy = 0.5f * (Y0 + Y1);
      anyCtr |= (ax > cx - 2.5f*w) && (ax < cx + 2.5f*w) &&
                (ay > cy - 2.5f*h) && (ay < cy + 2.5f*h);
    }
    fg_pen[r] = (anyBox || anyCtr) ? 0.0f : 10000.0f;
  }
  __syncthreads();

  // ---- wave phase: each wave = 16 query rows, sweeps 7 gt tiles ----
  const int wave = tid >> 5;
  const int lane = tid & 31;
  const int grp  = lane >> 4;   // 0/1 half of wave
  const int mn   = lane & 15;   // M for A-frag, N for B/C-frags
  const int qrow = wave * 16;

  for (int t = 0; t < GTILES; ++t) {
    const int gbase = t * 16;
    const int g = gbase + mn;
    short c1g = -1, c2g = -1, c3g = -1;
    if (g < GG) { c1g = gc1[g]; c2g = gc2[g]; c3g = gc3[g]; }

    CFrag acc; acc.v = (v8f){0,0,0,0,0,0,0,0};
    #pragma unroll
    for (int kc = 0; kc < 2; ++kc) {
      AFrag a, bm;
      #pragma unroll
      for (int e = 0; e < 16; ++e) {
        // 16-bit A/B fragment K mapping (ISA 7.12.2): VGPR i holds K pair;
        // lanes 0-15: K=0..7,16..23 ; lanes 16-31: K=8..15,24..31
        int k   = (e < 8) ? (grp * 8 + e) : (16 + grp * 8 + (e - 8));
        int cls = kc * 32 + k;
        a.h[e]  = F_lds[(qrow + mn) * KPAD + cls];
        bool on = (cls == c1g) || (cls == c2g) || (cls == c3g);
        bm.h[e] = on ? (_Float16)1.0f : (_Float16)0.0f;
      }
      acc.v = __builtin_amdgcn_wmma_f32_16x16x32_f16(
                  false, a.v, false, bm.v, (short)0, acc.v, false, false);
    }

    if (g < GG) {
      float X0 = gt_lds[g*4+0], Y0 = gt_lds[g*4+1], X1 = gt_lds[g*4+2], Y1 = gt_lds[g*4+3];
      float areaB = (X1 - X0) * (Y1 - Y0);
      float w = X1 - X0, h = Y1 - Y0, cx = 0.5f * (X0 + X1), cy = 0.5f * (Y0 + Y1);
      float nX0 = X0 / isz[0], nY0 = Y0 / isz[1], nX1 = X1 / isz[2], nY1 = Y1 / isz[3];
      #pragma unroll
      for (int r = 0; r < 8; ++r) {
        int M = r + grp * 8;             // C/D layout: VGPR r <-> M=r (lanes<16) / r+8
        int q = qbase + qrow + M;
        if (q >= QQ) continue;
        int rl = qrow + M;
        float x0 = pb_lds[rl*4+0], y0 = pb_lds[rl*4+1], x1 = pb_lds[rl*4+2], y1 = pb_lds[rl*4+3];
        float ax = 0.5f * (x0 + x1), ay = 0.5f * (y0 + y1);
        float areaA = (x1 - x0) * (y1 - y0);
        float iw = fmaxf(fminf(x1, X1) - fmaxf(x0, X0), 0.0f);
        float ih = fmaxf(fminf(y1, Y1) - fmaxf(y0, Y0), 0.0f);
        float inter = iw * ih;
        float uni   = areaA + areaB - inter;
        float iou   = inter / uni;
        float hw = fmaxf(fmaxf(x1, X1) - fminf(x0, X0), 0.0f);
        float hh = fmaxf(fmaxf(y1, Y1) - fminf(y0, Y0), 0.0f);
        float hull = hw * hh;
        float giou = iou - (hull - uni) / hull;
        float cb = fabsf(x0 / isz[0] - nX0) + fabsf(y0 / isz[1] - nY0) +
                   fabsf(x1 / isz[2] - nX1) + fabsf(y1 / isz[3] - nY1);
        bool ib = (ax > X0) && (ax < X1) && (ay > Y0) && (ay < Y1);
        bool ic = (ax > cx - 2.5f*w) && (ax < cx + 2.5f*w) &&
                  (ay > cy - 2.5f*h) && (ay < cy + 2.5f*h);
        float cost = cb + (1.0f / 3.0f) * acc.f[r] - giou +
                     ((ib && ic) ? 0.0f : 100.0f) + fg_pen[rl];
        costOut[(size_t)b * QQ * GG + (size_t)q * GG + g] = cost;
      }
    }
  }
}

// ============================================================
// Kernel 2: dynamic-k matching. 1 block/image, 512 threads.
// Loop-state trick: c_ only ever gets row-uniform +1e5 adds, so the mutable
// cost is cost[q,g] + 1e5*rowK[q]; row-argmins are invariant (precomputed).
// ============================================================
__global__ __launch_bounds__(512) void match_kernel(
    const float* __restrict__ pbox, const float* __restrict__ gbox,
    const float* __restrict__ costIn,
    float* __restrict__ mOut, float* __restrict__ selOut,
    float* __restrict__ gtOut, float* __restrict__ mqOut)
{
  __shared__ float gt_lds[GG * 4];
  __shared__ short dynk[GG];
  __shared__ short posQ[GG * 5];
  __shared__ int   rowCnt[QQ];        // 16 KB (LDS atomics)
  __shared__ short rowGt[QQ];         // 8 KB : -1 or matched gt
  __shared__ short rowAmin[QQ];       // 8 KB : argmin_g cost[q,:]
  __shared__ unsigned short rowK[QQ]; // 8 KB : c_ = cost + 1e5*rowK
  __shared__ int   colCnt[GG];
  __shared__ int   nUnmatched;

  const int b = blockIdx.x;
  const int tid = threadIdx.x;
  const float* __restrict__ costB = costIn + (size_t)b * QQ * GG;

  for (int i = tid; i < GG * 4; i += 512)
    gt_lds[i] = gbox[(size_t)b * GG * 4 + i];
  __syncthreads();

  // ---- per-gt: top-5 IoU sum -> dynamic k ; top-5 min-cost query indices ----
  for (int g = tid; g < GG; g += 512) {
    float X0 = gt_lds[g*4+0], Y0 = gt_lds[g*4+1], X1 = gt_lds[g*4+2], Y1 = gt_lds[g*4+3];
    float areaB = (X1 - X0) * (Y1 - Y0);
    float tv[5] = {-1e30f, -1e30f, -1e30f, -1e30f, -1e30f};           // ious, desc
    float cv[5] = {1e30f, 1e30f, 1e30f, 1e30f, 1e30f};                // costs, asc
    short ci[5] = {0, 0, 0, 0, 0};
    for (int q = 0; q < QQ; ++q) {
      const float* pq = pbox + (size_t)b * QQ * 4 + (size_t)q * 4;
      float x0 = pq[0], y0 = pq[1], x1 = pq[2], y1 = pq[3];
      float iw = fmaxf(fminf(x1, X1) - fmaxf(x0, X0), 0.0f);
      float ih = fmaxf(fminf(y1, Y1) - fmaxf(y0, Y0), 0.0f);
      float inter = iw * ih;
      float iou = inter / ((x1 - x0) * (y1 - y0) + areaB - inter);
      #pragma unroll
      for (int k = 0; k < 5; ++k)
        if (iou > tv[k]) {                              // insert (descending)
          for (int j = 4; j > k; --j) tv[j] = tv[j-1];
          tv[k] = iou; break;
        }
      __builtin_prefetch(&costB[(size_t)(q + 64) * GG + g], 0, 1);
      float cval = costB[(size_t)q * GG + g];
      #pragma unroll
      for (int k = 0; k < 5; ++k)
        if (cval < cv[k]) {                             // insert (ascending)
          for (int j = 4; j > k; --j) { cv[j] = cv[j-1]; ci[j] = ci[j-1]; }
          cv[k] = cval; ci[k] = (short)q; break;
        }
    }
    float s = tv[0] + tv[1] + tv[2] + tv[3] + tv[4];
    int dk = (int)s; if (dk < 1) dk = 1; if (dk > 5) dk = 5;
    dynk[g] = (short)dk;
    #pragma unroll
    for (int k = 0; k < 5; ++k) posQ[g * 5 + k] = ci[k];
  }

  // ---- per-row argmin (invariant under row-uniform additions) + init ----
  for (int q = tid; q < QQ; q += 512) {
    float best = 1e30f; int bg = 0;
    const float* row = costB + (size_t)q * GG;
    for (int g = 0; g < GG; ++g) {
      float v = row[g];
      if (v < best) { best = v; bg = g; }
    }
    rowAmin[q] = (short)bg;
    rowGt[q] = -1; rowCnt[q] = 0; rowK[q] = 0;
  }
  __syncthreads();

  // ---- initial scatter: m[posQ[g][k], g] = 1 for k < dynk[g] ----
  for (int g = tid; g < GG; g += 512)
    for (int k = 0; k < (int)dynk[g]; ++k) {
      int q = posQ[g * 5 + k];
      int old = atomicAdd(&rowCnt[q], 1);
      if (old == 0) rowGt[q] = (short)g;
    }
  __syncthreads();
  for (int q = tid; q < QQ; q += 512)                   // resolve
    if (rowCnt[q] > 1) { rowGt[q] = rowAmin[q]; rowCnt[q] = 1; }
  __syncthreads();

  // ---- while(any column unmatched) ----
  for (int iter = 0; iter < 300; ++iter) {
    for (int g = tid; g < GG; g += 512) colCnt[g] = 0;
    if (tid == 0) nUnmatched = 0;
    __syncthreads();
    for (int q = tid; q < QQ; q += 512)
      if (rowGt[q] >= 0) atomicAdd(&colCnt[rowGt[q]], 1);
    __syncthreads();
    for (int g = tid; g < GG; g += 512)
      if (colCnt[g] == 0) atomicAdd(&nUnmatched, 1);
    __syncthreads();
    if (nUnmatched == 0) break;

    for (int q = tid; q < QQ; q += 512)                 // c_ += 1e5 on matched rows
      if (rowGt[q] >= 0) rowK[q]++;
    __syncthreads();
    for (int g = tid; g < GG; g += 512)                 // claim unmatched columns
      if (colCnt[g] == 0) {
        float best = 1e30f; int bq = 0;
        for (int q = 0; q < QQ; ++q) {
          float v = costB[(size_t)q * GG + g] + 100000.0f * (float)rowK[q];
          if (v < best) { best = v; bq = q; }
        }
        int old = atomicAdd(&rowCnt[bq], 1);
        if (old == 0) rowGt[bq] = (short)g;
      }
    __syncthreads();
    for (int q = tid; q < QQ; q += 512)                 // resolve
      if (rowCnt[q] > 1) { rowGt[q] = rowAmin[q]; rowCnt[q] = 1; }
    __syncthreads();
  }

  // ---- outputs ----
  for (int i = tid; i < QQ * GG; i += 512) {
    int q = i / GG, g = i - q * GG;
    mOut[(size_t)b * QQ * GG + i] = (rowGt[q] == g) ? 1.0f : 0.0f;
  }
  for (int q = tid; q < QQ; q += 512) {
    selOut[b * QQ + q] = (rowGt[q] >= 0) ? 1.0f : 0.0f;
    gtOut[b * QQ + q]  = (rowGt[q] >= 0) ? (float)rowGt[q] : 0.0f;
  }
  for (int g = tid; g < GG; g += 512) {                 // argmin_q of c_ where m>0
    float best = 1e30f; int bq = 0;
    for (int q = 0; q < QQ; ++q)
      if (rowGt[q] == g) {
        float v = costB[(size_t)q * GG + g] + 100000.0f * (float)rowK[q];
        if (v < best) { best = v; bq = q; }
      }
    mqOut[b * GG + g] = (float)bq;
  }
}

// ============================================================
extern "C" void kernel_launch(void* const* d_in, const int* in_sizes, int n_in,
                              void* d_out, int out_size, void* d_ws, size_t ws_size,
                              hipStream_t stream) {
  (void)in_sizes; (void)n_in; (void)out_size; (void)d_ws; (void)ws_size;
  const float* pl1  = (const float*)d_in[0];
  const float* pl2  = (const float*)d_in[1];
  const float* pl3  = (const float*)d_in[2];
  const float* pbox = (const float*)d_in[3];
  const float* gbox = (const float*)d_in[4];
  const float* isz  = (const float*)d_in[5];
  const int*   lab1 = (const int*)d_in[6];
  const int*   lab2 = (const int*)d_in[7];
  const int*   lab3 = (const int*)d_in[8];
  float* out = (float*)d_out;

  const size_t OFF_SEL  = (size_t)BB * QQ * GG;
  const size_t OFF_GT   = OFF_SEL + (size_t)BB * QQ;
  const size_t OFF_MQ   = OFF_GT + (size_t)BB * QQ;
  const size_t OFF_COST = OFF_MQ + (size_t)BB * GG;

  dim3 grid1((QQ + QTILE - 1) / QTILE, BB);
  cost_kernel<<<grid1, 256, 0, stream>>>(pl1, pl2, pl3, pbox, gbox, isz,
                                         lab1, lab2, lab3, out + OFF_COST);
  match_kernel<<<BB, 512, 0, stream>>>(pbox, gbox, out + OFF_COST,
                                       out, out + OFF_SEL, out + OFF_GT, out + OFF_MQ);
}
